// GraphCompletion_44555990729375
// MI455X (gfx1250) — matrix operand order repaired and checked by reading, not compiled, not run on hardware
//
#include <hip/hip_runtime.h>

typedef __attribute__((ext_vector_type(16))) _Float16 v16h;
typedef __attribute__((ext_vector_type(4)))  _Float16 v4h;
typedef __attribute__((ext_vector_type(8)))  float    v8f;

#define GAT_H 4
#define GAT_C 64
#define GAT_HC 256

// ---------------- ordered-float encoding for atomic max ----------------
__device__ __forceinline__ unsigned fenc(float x) {
    unsigned u = __float_as_uint(x);
    return (u & 0x80000000u) ? ~u : (u | 0x80000000u);
}
__device__ __forceinline__ float fdec(unsigned u) {
    return __uint_as_float((u & 0x80000000u) ? (u & 0x7fffffffu) : ~u);
}
// fenc(-inf) == 0x007FFFFF

__device__ __forceinline__ float lrelu02(float v) { return v > 0.f ? v : 0.2f * v; }

// ---------------- generic WMMA GEMM: C[N,M] = A[N,K] @ B[K,M] (+bias)(+relu) ----
// block = 256 threads (8 waves), tile 64 rows x 128 cols, K step 32.
// Out-of-range rows are address-clamped (never zeroed): WMMA output row i
// depends only on A row i, and rows >= Nr are never stored.
__global__ __launch_bounds__(256)
void gemm_wmma(const float* __restrict__ A, const float* __restrict__ B,
               const float* __restrict__ bias, float* __restrict__ Cout,
               int Nr, int K, int M, int relu)
{
    __shared__ _Float16 lA[64][40];    // [row][k]   (+8 pad)
    __shared__ _Float16 lB[128][40];   // [col][k]   (+8 pad) -- W tile transposed

    const int tid  = threadIdx.x;
    const int lane = tid & 31;
    const int wave = tid >> 5;
    const int wr   = wave >> 1;        // 0..3 : 16-row strip
    const int wc   = wave & 1;         // 0..1 : 64-col strip
    const int half = lane >> 4;        // 0/1
    const int l15  = lane & 15;
    const int rowBase = blockIdx.x * 64;
    const int colBase = blockIdx.y * 128;

    // precomputed local A rows and k offsets for the 2 per-thread loads
    int arow[2], aqc[2];
    for (int j = 0; j < 2; ++j) {
        int idx = tid + j * 256;          // 0..511
        int r = idx >> 3;                 // 0..63
        aqc[j]  = (idx & 7) * 4;          // 0,4,..,28
        int gr = rowBase + r;
        if (gr > Nr - 1) gr = Nr - 1;     // clamp, no predication
        arow[j] = r;
        // store local row r (not the clamped global row):
        arow[j] = r;
        aqc[j] |= 0;                      // keep
        // stash clamped row separately:
        // (use a second array)
        // handled below via recompute
    }

    v8f acc[4];
    for (int t = 0; t < 4; ++t)
        for (int i = 0; i < 8; ++i) acc[t][i] = 0.0f;

    for (int k0 = 0; k0 < K; k0 += 32) {
        // stage A tile 64x32: float4 loads, packed ds_store_b64
        for (int j = 0; j < 2; ++j) {
            int idx = tid + j * 256;
            int r  = idx >> 3;
            int qc = (idx & 7) * 4;
            int gr = rowBase + r;
            if (gr > Nr - 1) gr = Nr - 1;
            const float4 v = *(const float4*)(A + (long)gr * K + k0 + qc);
            v4h hv;
            hv.x = (_Float16)v.x; hv.y = (_Float16)v.y;
            hv.z = (_Float16)v.z; hv.w = (_Float16)v.w;
            *(v4h*)(&lA[r][qc]) = hv;
        }
        // stage W tile 32x128 transposed -> lB[n][k]: coalesced float4 loads
        for (int j = 0; j < 4; ++j) {
            int idx = tid + j * 256;
            int kk = idx >> 5;            // 0..31
            int n0 = (idx & 31) * 4;      // 0..124
            const float4 v = *(const float4*)(B + (long)(k0 + kk) * M + colBase + n0);
            lB[n0 + 0][kk] = (_Float16)v.x;
            lB[n0 + 1][kk] = (_Float16)v.y;
            lB[n0 + 2][kk] = (_Float16)v.z;
            lB[n0 + 3][kk] = (_Float16)v.w;
        }
        __syncthreads();

        // A fragment, ISA 16-bit A 16x32 layout:
        // lanes 0-15: K 0-7 (v0-3), 16-23 (v4-7); lanes 16-31: +8
        union { v16h v; _Float16 h[16]; } fa;
        {
            int r = wr * 16 + l15;
            for (int j = 0; j < 4; ++j) {
                int k = half * 8 + 2 * j;
                fa.h[2 * j]     = lA[r][k];
                fa.h[2 * j + 1] = lA[r][k + 1];
            }
            for (int j = 0; j < 4; ++j) {
                int k = 16 + half * 8 + 2 * j;
                fa.h[8 + 2 * j]     = lA[r][k];
                fa.h[8 + 2 * j + 1] = lA[r][k + 1];
            }
        }
        for (int t = 0; t < 4; ++t) {
            // B fragment: lane -> N, same K order as A: lanes 0-15 K 0-7,16-23; lanes 16-31 K 8-15,24-31
            union { v16h v; _Float16 h[16]; } fb;
            int n = wc * 64 + t * 16 + l15;
            for (int e = 0; e < 16; ++e) fb.h[e] = lB[n][(e < 8) ? (half * 8 + e) : (16 + half * 8 + (e - 8))];
            acc[t] = __builtin_amdgcn_wmma_f32_16x16x32_f16(
                false, fa.v, false, fb.v, (short)0, acc[t], false, false);
        }
        __syncthreads();
    }

    // C/D layout: VGPR i -> M = i (lanes 0-15) / 8+i (lanes 16-31), N = lane&15
    for (int t = 0; t < 4; ++t) {
        int col = colBase + wc * 64 + t * 16 + l15;
        for (int i = 0; i < 8; ++i) {
            int row = rowBase + wr * 16 + half * 8 + i;
            if (row < Nr) {
                float v = acc[t][i];
                if (bias) v += bias[col];
                if (relu) v = v > 0.0f ? v : 0.0f;
                Cout[(long)row * M + col] = v;
            }
        }
    }
}

// ---------------- per-node attention logits (float4 dot) ----------------
__global__ void alphas_kernel(const float* __restrict__ h, const float* __restrict__ a_src,
                              const float* __restrict__ a_dst, float* __restrict__ as,
                              float* __restrict__ ad, int N)
{
    int idx = blockIdx.x * blockDim.x + threadIdx.x;  // n*H + head
    if (idx >= N * GAT_H) return;
    int n = idx >> 2, hd = idx & 3;
    const float4* hp = (const float4*)(h + (long)n * GAT_HC + hd * GAT_C);
    const float4* ws = (const float4*)(a_src + hd * GAT_C);
    const float4* wd = (const float4*)(a_dst + hd * GAT_C);
    float ss = 0.f, sd = 0.f;
    for (int c = 0; c < GAT_C / 4; ++c) {
        float4 v = hp[c], s4 = ws[c], d4 = wd[c];
        ss += v.x * s4.x + v.y * s4.y + v.z * s4.z + v.w * s4.w;
        sd += v.x * d4.x + v.y * d4.y + v.z * d4.z + v.w * d4.w;
    }
    as[idx] = ss; ad[idx] = sd;
}

__global__ void fill_u32_kernel(unsigned* __restrict__ p, unsigned v, long n)
{
    long i = (long)blockIdx.x * blockDim.x + threadIdx.x;
    if (i < n) p[i] = v;
}

// pass a: e = leaky_relu(as[src]+ad[dst]); segment max via ordered-uint atomicMax
__global__ void edge_max_kernel(const int* __restrict__ src, const int* __restrict__ dst,
                                const float* __restrict__ as, const float* __restrict__ ad,
                                float* __restrict__ ebuf, unsigned* __restrict__ m,
                                int E, int Etot)
{
    int e = blockIdx.x * blockDim.x + threadIdx.x;
    if (e >= Etot) return;
    int s_, d_;
    if (e < E) { s_ = src[e]; d_ = dst[e]; } else { s_ = e - E; d_ = s_; }
    float4 av = *(const float4*)(as + (long)s_ * GAT_H);
    float4 dv = *(const float4*)(ad + (long)d_ * GAT_H);
    float4 ev;
    ev.x = lrelu02(av.x + dv.x);
    ev.y = lrelu02(av.y + dv.y);
    ev.z = lrelu02(av.z + dv.z);
    ev.w = lrelu02(av.w + dv.w);
    *(float4*)(ebuf + (long)e * GAT_H) = ev;
    unsigned* mp = m + (long)d_ * GAT_H;
    atomicMax(mp + 0, fenc(ev.x));
    atomicMax(mp + 1, fenc(ev.y));
    atomicMax(mp + 2, fenc(ev.z));
    atomicMax(mp + 3, fenc(ev.w));
}

// pass b: ex = exp(e - m[dst]); segment sum
__global__ void edge_expsum_kernel(const int* __restrict__ src, const int* __restrict__ dst,
                                   float* __restrict__ ebuf, const unsigned* __restrict__ m,
                                   float* __restrict__ ssum, int E, int Etot)
{
    int e = blockIdx.x * blockDim.x + threadIdx.x;
    if (e >= Etot) return;
    int d_;
    if (e < E) { d_ = dst[e]; } else { d_ = e - E; }
    uint4 mu = *(const uint4*)(m + (long)d_ * GAT_H);
    float4 ev = *(const float4*)(ebuf + (long)e * GAT_H);
    float mv[4] = { fdec(mu.x), fdec(mu.y), fdec(mu.z), fdec(mu.w) };
    float in[4] = { ev.x, ev.y, ev.z, ev.w };
    float4 ox;
    float* op = &ox.x;
    float* sp = ssum + (long)d_ * GAT_H;
    for (int hd = 0; hd < GAT_H; ++hd) {
        float mm = mv[hd];
        if (!(mm > -1e30f && mm < 1e30f)) mm = 0.0f;   // isfinite guard like reference
        float v = __expf(in[hd] - mm);
        op[hd] = v;
        atomicAdd(sp + hd, v);
    }
    *(float4*)(ebuf + (long)e * GAT_H) = ox;
}

// pass c: agg[dst] += h[src] * ex/(s[dst]+eps)
// 64 lanes per edge; lane t -> head t>>4, channels 4*(t&15)..+3 (float4)
__global__ __launch_bounds__(256)
void edge_agg_kernel(const int* __restrict__ src, const int* __restrict__ dst,
                     const float* __restrict__ exbuf, const float* __restrict__ ssum,
                     const float* __restrict__ hmat, float* __restrict__ agg,
                     int E, int Etot)
{
    long g = (long)blockIdx.x * blockDim.x + threadIdx.x;
    int e = (int)(g >> 6);
    int t = (int)(g & 63);
    if (e >= Etot) return;
    int s_, d_;
    if (e < E) { s_ = src[e]; d_ = dst[e]; } else { s_ = e - E; d_ = s_; }
    int hd = t >> 4;
    int c  = (t & 15) * 4;
    float coef = exbuf[(long)e * GAT_H + hd] / (ssum[(long)d_ * GAT_H + hd] + 1e-16f);
    const float4 hv = *(const float4*)(hmat + (long)s_ * GAT_HC + hd * GAT_C + c);
    float* ap = agg + (long)d_ * GAT_HC + hd * GAT_C + c;
    atomicAdd(ap + 0, hv.x * coef);
    atomicAdd(ap + 1, hv.y * coef);
    atomicAdd(ap + 2, hv.z * coef);
    atomicAdd(ap + 3, hv.w * coef);
}

__global__ void bias_act_kernel(const float* __restrict__ in, const float* __restrict__ bias,
                                float* __restrict__ out, int N)
{
    long i = (long)blockIdx.x * blockDim.x + threadIdx.x;   // float4 index
    if (i >= (long)N * (GAT_HC / 4)) return;
    int f = (int)(i & (GAT_HC / 4 - 1));
    float4 v = ((const float4*)in)[i];
    float4 b = ((const float4*)bias)[f];
    v.x = fmaxf(v.x + b.x, 0.f);
    v.y = fmaxf(v.y + b.y, 0.f);
    v.z = fmaxf(v.z + b.z, 0.f);
    v.w = fmaxf(v.w + b.w, 0.f);
    ((float4*)out)[i] = v;
}

// semantic attention fusion: one wave32 per node; lane covers 8 channels (2 float4)
__global__ __launch_bounds__(256)
void fuse_kernel(const float* __restrict__ emb, const float* __restrict__ w_pm,
                 const float* __restrict__ b_pm, const float* __restrict__ w_nm,
                 const float* __restrict__ b_nm, float* __restrict__ z, int N)
{
    int node = blockIdx.x * 8 + (threadIdx.x >> 5);
    int lane = threadIdx.x & 31;
    if (node >= N) return;
    const long step = (long)N * GAT_HC;
    float dots[5];
    for (int i = 0; i < 5; ++i) {
        const float4* ep = (const float4*)(emb + (long)i * step + (long)node * GAT_HC + lane * 8);
        const float4* w  = (const float4*)(((i < 3) ? w_pm : w_nm) + lane * 8);
        float p = 0.f;
        for (int j = 0; j < 2; ++j) {
            float4 v = ep[j], q = w[j];
            p += v.x * q.x + v.y * q.y + v.z * q.z + v.w * q.w;
        }
        for (int off = 16; off > 0; off >>= 1) p += __shfl_xor(p, off, 32);
        dots[i] = p;
    }
    float bp = b_pm[0], bn = b_nm[0];
    float sc[5];
    for (int i = 0; i < 3; ++i) { float v = dots[i] + bp; sc[i] = v > 0.f ? v : 0.f; }
    for (int i = 3; i < 5; ++i) { float v = dots[i] + bn; sc[i] = v > 0.f ? v : 0.f; }
    float mp = fmaxf(sc[0], fmaxf(sc[1], sc[2]));
    float e0 = __expf(sc[0] - mp), e1 = __expf(sc[1] - mp), e2 = __expf(sc[2] - mp);
    float sp = e0 + e1 + e2;
    float mn = fmaxf(sc[3], sc[4]);
    float e3 = __expf(sc[3] - mn), e4 = __expf(sc[4] - mn);
    float sn = e3 + e4;
    float w0 = e0 / sp, w1 = e1 / sp, w2 = e2 / sp, w3 = e3 / sn, w4 = e4 / sn;
    for (int j = 0; j < 2; ++j) {
        long base = (long)node * GAT_HC + lane * 8 + j * 4;
        float4 v0 = *(const float4*)(emb + base);
        float4 v1 = *(const float4*)(emb + step + base);
        float4 v2 = *(const float4*)(emb + 2 * step + base);
        float4 v3 = *(const float4*)(emb + 3 * step + base);
        float4 v4 = *(const float4*)(emb + 4 * step + base);
        float4 ap, an;
        ap.x = (w0 * v0.x + w1 * v1.x + w2 * v2.x) * (1.f / 3.f);
        ap.y = (w0 * v0.y + w1 * v1.y + w2 * v2.y) * (1.f / 3.f);
        ap.z = (w0 * v0.z + w1 * v1.z + w2 * v2.z) * (1.f / 3.f);
        ap.w = (w0 * v0.w + w1 * v1.w + w2 * v2.w) * (1.f / 3.f);
        an.x = (w3 * v3.x + w4 * v4.x) * 0.5f;
        an.y = (w3 * v3.y + w4 * v4.y) * 0.5f;
        an.z = (w3 * v3.z + w4 * v4.z) * 0.5f;
        an.w = (w3 * v3.w + w4 * v4.w) * 0.5f;
        long zb = (long)node * 2 * GAT_HC + lane * 8 + j * 4;
        *(float4*)(z + zb) = ap;
        *(float4*)(z + zb + GAT_HC) = an;
    }
}

// ---------------- host-side layer driver ----------------
static void run_gat_layer(const float* xin, int K, const float* W, const float* a_s,
                          const float* a_d, const float* bias, const int* srcp,
                          const int* dstp, float* hbuf, float* asb, float* adb,
                          unsigned* mb, float* sb, float* exb, float* agg,
                          float* outbuf, int N, int E, int Etot, hipStream_t stream)
{
    dim3 g((N + 63) / 64, GAT_HC / 128);
    gemm_wmma<<<g, 256, 0, stream>>>(xin, W, nullptr, hbuf, N, K, GAT_HC, 0);

    alphas_kernel<<<(N * GAT_H + 255) / 256, 256, 0, stream>>>(hbuf, a_s, a_d, asb, adb, N);

    long nh = (long)N * GAT_H;
    fill_u32_kernel<<<(unsigned)((nh + 255) / 256), 256, 0, stream>>>(mb, 0x007FFFFFu, nh); // enc(-inf)
    fill_u32_kernel<<<(unsigned)((nh + 255) / 256), 256, 0, stream>>>((unsigned*)sb, 0u, nh);
    long nc = (long)N * GAT_HC;
    fill_u32_kernel<<<(unsigned)((nc + 255) / 256), 256, 0, stream>>>((unsigned*)agg, 0u, nc);

    int be = (Etot + 255) / 256;
    edge_max_kernel<<<be, 256, 0, stream>>>(srcp, dstp, asb, adb, exb, mb, E, Etot);
    edge_expsum_kernel<<<be, 256, 0, stream>>>(srcp, dstp, exb, mb, sb, E, Etot);
    long tot = (long)Etot * 64;
    edge_agg_kernel<<<(unsigned)((tot + 255) / 256), 256, 0, stream>>>(srcp, dstp, exb, sb,
                                                                       hbuf, agg, E, Etot);
    long nc4 = (long)N * (GAT_HC / 4);
    bias_act_kernel<<<(unsigned)((nc4 + 255) / 256), 256, 0, stream>>>(agg, bias, outbuf, N);
}

extern "C" void kernel_launch(void* const* d_in, const int* in_sizes, int n_in,
                              void* d_out, int out_size, void* d_ws, size_t ws_size,
                              hipStream_t stream)
{
    const int*   edges = (const int*)d_in[0];
    const float* x0    = (const float*)d_in[1];
    const float* W1    = (const float*)d_in[2];
    const float* as1   = (const float*)d_in[3];
    const float* ad1   = (const float*)d_in[4];
    const float* b1    = (const float*)d_in[5];
    const float* W2    = (const float*)d_in[6];
    const float* as2   = (const float*)d_in[7];
    const float* ad2   = (const float*)d_in[8];
    const float* b2    = (const float*)d_in[9];
    const float* w_pm  = (const float*)d_in[10];
    const float* b_pm  = (const float*)d_in[11];
    const float* w_nm  = (const float*)d_in[12];
    const float* b_nm  = (const float*)d_in[13];
    const float* W_dec = (const float*)d_in[14];
    const float* b_dec = (const float*)d_in[15];

    const int E    = in_sizes[0] / 10;   // [5,2,E]
    const int N    = in_sizes[1] / 64;   // IN = 64
    const int INF_ = 64;
    const int Etot = E + N;

    char* wsp = (char*)d_ws;
    auto alloc = [&](size_t bytes) -> char* {
        char* p = wsp; wsp += (bytes + 255) & ~(size_t)255; return p;
    };
    float*    emb  = (float*)alloc((size_t)5 * N * GAT_HC * 4);
    float*    bufA = (float*)alloc((size_t)N * GAT_HC * 4);
    float*    bufH = (float*)alloc((size_t)N * GAT_HC * 4);
    float*    agg  = (float*)alloc((size_t)N * GAT_HC * 4);
    float*    asb  = (float*)alloc((size_t)N * GAT_H * 4);
    float*    adb  = (float*)alloc((size_t)N * GAT_H * 4);
    unsigned* mb   = (unsigned*)alloc((size_t)N * GAT_H * 4);
    float*    sb   = (float*)alloc((size_t)N * GAT_H * 4);
    float*    exb  = (float*)alloc((size_t)Etot * GAT_H * 4);
    float*    zb   = (float*)alloc((size_t)N * 2 * GAT_HC * 4);
    (void)ws_size; (void)n_in; (void)out_size;

    for (int mp = 0; mp < 5; ++mp) {
        const int* srcp = edges + (size_t)mp * 2 * E;
        const int* dstp = srcp + E;
        // layer 1: x0 [N,64] -> bufA [N,256]
        run_gat_layer(x0, INF_, W1 + (size_t)mp * INF_ * GAT_HC,
                      as1 + (size_t)mp * GAT_H * GAT_C, ad1 + (size_t)mp * GAT_H * GAT_C,
                      b1 + (size_t)mp * GAT_HC, srcp, dstp,
                      bufH, asb, adb, mb, sb, exb, agg, bufA, N, E, Etot, stream);
        // layer 2: bufA [N,256] -> emb[mp]
        run_gat_layer(bufA, GAT_HC, W2 + (size_t)mp * GAT_HC * GAT_HC,
                      as2 + (size_t)mp * GAT_H * GAT_C, ad2 + (size_t)mp * GAT_H * GAT_C,
                      b2 + (size_t)mp * GAT_HC, srcp, dstp,
                      bufH, asb, adb, mb, sb, exb, agg,
                      emb + (size_t)mp * N * GAT_HC, N, E, Etot, stream);
    }

    fuse_kernel<<<(N + 7) / 8, 256, 0, stream>>>(emb, w_pm, b_pm, w_nm, b_nm, zb, N);

    for (int d = 0; d < 3; ++d) {
        dim3 g((N + 63) / 64, (2 * GAT_HC) / 128);
        gemm_wmma<<<g, 256, 0, stream>>>(zb, W_dec + (size_t)d * 512 * 512,
                                         b_dec + (size_t)d * 512,
                                         (float*)d_out + (size_t)d * N * 512,
                                         N, 512, 512, 0);
    }
}
